// SSM8Block_68685116998087
// MI455X (gfx1250) — compile-verified, matrix operand order checked by reading, not run on hardware
//
#include <hip/hip_runtime.h>
#include <hip/hip_bf16.h>
#include <math.h>

// ---------------------------------------------------------------------------
// Mamba-style SSM block for MI455X (gfx1250, wave32, WMMA).
// GEMMs: v_wmma_f32_16x16x32_bf16, wave tile 32x64 (8 WMMA / k-step).
// Scan: per-(b,channel) lane, 32-state vector in VGPRs, B_t/C_t staged via
// double-buffered GLOBAL_LOAD_ASYNC_TO_LDS_B64 (ASYNCcnt) one step ahead.
// ---------------------------------------------------------------------------

#define D_MODEL 1024
#define D_INNER 2048
#define D_STATE 32
#define D_CONV  4
#define DT_RANK 64
#define BSZ     2
#define SEQ     2048
#define NTOK    (BSZ * SEQ)              // 4096 tokens
#define XPROJ_N (DT_RANK + 2 * D_STATE)  // 128

typedef __bf16 bf16_t;
typedef __attribute__((ext_vector_type(16))) __bf16 v16bf;
typedef __attribute__((ext_vector_type(8)))  __bf16 v8bf;
typedef __attribute__((ext_vector_type(8)))  float  v8f;

// LDS byte-offset of a shared-memory pointer (ptrtoint of addrspace(3)).
#define LDS_OFF(p) ((unsigned)(unsigned long long)(__attribute__((address_space(3))) const void*)(p))

// Load one 16x32 bf16 fragment row for this lane.
// Layout (ISA 05_wmma, 16-bit A 16x32): lanes 0-15 hold K {kk+0..7, kk+16..23},
// lanes 16-31 hold K {kk+8..15, kk+24..31}; ko already includes the half shift.
__device__ __forceinline__ v16bf load_frag32(const bf16_t* __restrict__ rowp, int ko) {
    v8bf lo = *(const v8bf*)(rowp + ko);
    v8bf hi = *(const v8bf*)(rowp + ko + 16);
    return __builtin_shufflevector(lo, hi, 0, 1, 2, 3, 4, 5, 6, 7,
                                           8, 9, 10, 11, 12, 13, 14, 15);
}

// ---------------------------------------------------------------------------
// GEMM: out[M,N] = A[M,K] (bf16, row-major) x W[N,K]^T (bf16, row-major).
// Wave tile 32(M) x 64(N): 2 A-fragments x 4 B-fragments -> 8 WMMA / k-step;
// B fragments are reused across both A fragments. Block = 8 waves stacked in
// M -> 256x64 block tile.
// mode 0: split columns into x_in (out0) / z (out1), each [*, D_INNER]
// mode 1/3: plain f32 store to out0[row*N + col]
// mode 2: softplus(acc + bias0[col] + bias1[col]) -> out0 (delta path)
// ---------------------------------------------------------------------------
__global__ __launch_bounds__(256)
void gemm_wmma_bf16(const bf16_t* __restrict__ A, const bf16_t* __restrict__ W,
                    int K, int N, int mode,
                    const float* __restrict__ bias0, const float* __restrict__ bias1,
                    float* __restrict__ out0, float* __restrict__ out1) {
    const int lane    = threadIdx.x & 31;
    const int wave    = threadIdx.x >> 5;
    const int rowBase = blockIdx.x * 256 + wave * 32;
    const int colBase = blockIdx.y * 64;
    const int half    = lane >> 4;   // 0 | 1 (selects K-half per ISA layout)
    const int lrow    = lane & 15;   // M (for A) / N (for B) within tile

    const bf16_t* __restrict__ aRow0 = A + (size_t)(rowBase +  0 + lrow) * K;
    const bf16_t* __restrict__ aRow1 = A + (size_t)(rowBase + 16 + lrow) * K;
    const bf16_t* __restrict__ w0    = W + (size_t)(colBase +  0 + lrow) * K;
    const bf16_t* __restrict__ w1    = W + (size_t)(colBase + 16 + lrow) * K;
    const bf16_t* __restrict__ w2    = W + (size_t)(colBase + 32 + lrow) * K;
    const bf16_t* __restrict__ w3    = W + (size_t)(colBase + 48 + lrow) * K;

    v8f acc[2][4];
    #pragma unroll
    for (int s = 0; s < 2; ++s)
        #pragma unroll
        for (int j = 0; j < 4; ++j)
            acc[s][j] = (v8f){};

    for (int kk = 0; kk < K; kk += 32) {
        const int ko = kk + (half << 3);
        v16bf a0 = load_frag32(aRow0, ko);
        v16bf a1 = load_frag32(aRow1, ko);
        v16bf b0 = load_frag32(w0, ko);
        v16bf b1 = load_frag32(w1, ko);
        v16bf b2 = load_frag32(w2, ko);
        v16bf b3 = load_frag32(w3, ko);
        acc[0][0] = __builtin_amdgcn_wmma_f32_16x16x32_bf16(false, a0, false, b0, (short)0, acc[0][0], false, false);
        acc[0][1] = __builtin_amdgcn_wmma_f32_16x16x32_bf16(false, a0, false, b1, (short)0, acc[0][1], false, false);
        acc[0][2] = __builtin_amdgcn_wmma_f32_16x16x32_bf16(false, a0, false, b2, (short)0, acc[0][2], false, false);
        acc[0][3] = __builtin_amdgcn_wmma_f32_16x16x32_bf16(false, a0, false, b3, (short)0, acc[0][3], false, false);
        acc[1][0] = __builtin_amdgcn_wmma_f32_16x16x32_bf16(false, a1, false, b0, (short)0, acc[1][0], false, false);
        acc[1][1] = __builtin_amdgcn_wmma_f32_16x16x32_bf16(false, a1, false, b1, (short)0, acc[1][1], false, false);
        acc[1][2] = __builtin_amdgcn_wmma_f32_16x16x32_bf16(false, a1, false, b2, (short)0, acc[1][2], false, false);
        acc[1][3] = __builtin_amdgcn_wmma_f32_16x16x32_bf16(false, a1, false, b3, (short)0, acc[1][3], false, false);
    }

    // C/D layout: lane%16 = N, VGPR r = M r (lanes 0-15) / r+8 (lanes 16-31).
    #pragma unroll
    for (int s = 0; s < 2; ++s) {
        #pragma unroll
        for (int j = 0; j < 4; ++j) {
            v8f c = acc[s][j];
            const int col = colBase + j * 16 + lrow;
            #pragma unroll
            for (int r = 0; r < 8; ++r) {
                const int row = rowBase + s * 16 + r + half * 8;
                float v = c[r];
                if (mode == 0) {
                    if (col < D_INNER) out0[(size_t)row * D_INNER + col] = v;
                    else               out1[(size_t)row * D_INNER + (col - D_INNER)] = v;
                } else if (mode == 2) {
                    float xv = v + bias0[col] + bias1[col];
                    float sp = (xv > 20.0f) ? xv : log1pf(__expf(xv));  // softplus
                    out0[(size_t)row * N + col] = sp;
                } else {
                    out0[(size_t)row * N + col] = v;
                }
            }
        }
    }
}

// ---------------------------------------------------------------------------
// RMSNorm over D_MODEL per token -> bf16 activations for in_proj WMMA.
// ---------------------------------------------------------------------------
__global__ __launch_bounds__(256)
void rmsnorm_kernel(const float* __restrict__ x, const float* __restrict__ w,
                    bf16_t* __restrict__ out) {
    __shared__ float red[256];
    const int row = blockIdx.x;
    const float* __restrict__ xr = x + (size_t)row * D_MODEL;
    float s = 0.0f;
    #pragma unroll
    for (int i = 0; i < 4; ++i) {
        float v = xr[threadIdx.x + i * 256];
        s += v * v;
    }
    red[threadIdx.x] = s;
    __syncthreads();
    for (int off = 128; off > 0; off >>= 1) {
        if (threadIdx.x < off) red[threadIdx.x] += red[threadIdx.x + off];
        __syncthreads();
    }
    const float inv = rsqrtf(red[0] * (1.0f / D_MODEL) + 1e-5f);
    #pragma unroll
    for (int i = 0; i < 4; ++i) {
        const int c = threadIdx.x + i * 256;
        out[(size_t)row * D_MODEL + c] = (bf16_t)(xr[c] * inv * w[c]);
    }
}

// f32 -> bf16 bulk conversion (weights; re-done every launch, deterministic).
__global__ __launch_bounds__(256)
void convert_f32_bf16(const float* __restrict__ src, bf16_t* __restrict__ dst, int n) {
    const int i = blockIdx.x * blockDim.x + threadIdx.x;
    if (i < n) dst[i] = (bf16_t)src[i];
}

// dtr = xbc[:, 0:64] -> bf16 compacted [NTOK, 64] for the dt_proj WMMA GEMM.
__global__ __launch_bounds__(256)
void extract_dtr(const float* __restrict__ xbc, bf16_t* __restrict__ dtr) {
    const int i = blockIdx.x * blockDim.x + threadIdx.x;  // NTOK*DT_RANK
    const int row = i >> 6;
    const int c   = i & 63;
    dtr[i] = (bf16_t)xbc[(size_t)row * XPROJ_N + c];
}

// ---------------------------------------------------------------------------
// Depthwise causal conv (K=4, zero left-pad 3), one thread per (b, channel),
// in-place on x (f32) with shift registers; also emits bf16 copy for x_proj.
// ---------------------------------------------------------------------------
__global__ __launch_bounds__(256)
void conv1d_kernel(float* __restrict__ x, const float* __restrict__ conv_w,
                   const float* __restrict__ conv_b, bf16_t* __restrict__ xc_bf16) {
    const int gid = blockIdx.x * blockDim.x + threadIdx.x;   // 0..BSZ*D_INNER-1
    const int d = gid & (D_INNER - 1);
    const int b = gid >> 11;
    const float w0 = conv_w[d * 4 + 0], w1 = conv_w[d * 4 + 1];
    const float w2 = conv_w[d * 4 + 2], w3 = conv_w[d * 4 + 3];
    const float bias = conv_b[d];
    float x3 = 0.0f, x2 = 0.0f, x1 = 0.0f;
    const size_t base = (size_t)b * SEQ * D_INNER + d;
    for (int t = 0; t < SEQ; ++t) {
        const size_t idx = base + (size_t)t * D_INNER;
        const float xt = x[idx];
        const float o  = w0 * x3 + w1 * x2 + w2 * x1 + w3 * xt + bias;
        x[idx]       = o;             // f32 for scan / skip
        xc_bf16[idx] = (bf16_t)o;     // bf16 for x_proj WMMA
        x3 = x2; x2 = x1; x1 = xt;
    }
}

// ---------------------------------------------------------------------------
// Selective scan: one thread per (b, d). 32-state h vector + A row in VGPRs.
// Each wave prefetches the next step's B_t/C_t row (64 floats, wave-uniform)
// into its private double-buffered LDS slot with GLOBAL_LOAD_ASYNC_TO_LDS_B64
// (ASYNCcnt); s_wait_asynccnt 1 guarantees the current buffer is resident
// (async loads complete in order). Per-wave slots -> no barriers.
// Fuses skip (+u*D) and SiLU(z) gating; writes bf16 for out_proj.
// ---------------------------------------------------------------------------
__global__ __launch_bounds__(256)
void scan_kernel(const float* __restrict__ delta, const float* __restrict__ u,
                 const float* __restrict__ z, const float* __restrict__ xbc,
                 const float* __restrict__ A_log, const float* __restrict__ Dvec,
                 bf16_t* __restrict__ ygated) {
    __shared__ float lds_bc[8][2][64];   // [wave][buffer][B(32) | C(32)]

    const int gid  = blockIdx.x * blockDim.x + threadIdx.x;  // 0..BSZ*D_INNER-1
    const int lane = threadIdx.x & 31;
    const int wave = threadIdx.x >> 5;
    const int d = gid & (D_INNER - 1);
    const int b = gid >> 11;   // all 256 threads of a block share the same b

    float Aa[D_STATE], h[D_STATE];
    #pragma unroll
    for (int n = 0; n < D_STATE; ++n) {
        Aa[n] = -__expf(A_log[d * D_STATE + n]);
        h[n]  = 0.0f;
    }
    const float Dd = Dvec[d];
    const int rowBase = b * SEQ;

    // Issue async prefetch of row `r`'s B/C (64 floats) into buffer `buf`:
    // 32 lanes x 8B = 256B in one async instruction.
    auto issue_bc = [&](int r, int buf) {
        unsigned lds_off = LDS_OFF(&lds_bc[wave][buf][0]) + (unsigned)lane * 8u;
        unsigned long long gaddr =
            (unsigned long long)(xbc + (size_t)r * XPROJ_N + DT_RANK) +
            (unsigned long long)lane * 8ull;
        asm volatile("global_load_async_to_lds_b64 %0, %1, off"
                     :: "v"(lds_off), "v"(gaddr) : "memory");
    };

    issue_bc(rowBase, 0);   // prefetch t=0

    for (int t = 0; t < SEQ; ++t) {
        const int row = rowBase + t;
        const int nxt = (t + 1 < SEQ) ? row + 1 : row;  // clamp final prefetch
        issue_bc(nxt, (t + 1) & 1);
        // Outstanding: load(t), load(t+1). Wait <=1 -> load(t) complete.
        asm volatile("s_wait_asynccnt 0x1" ::: "memory");

        const size_t cidx = (size_t)row * D_INNER + d;
        const float dt = delta[cidx];
        const float ut = u[cidx];
        const float zt = z[cidx];

        const float* __restrict__ bc = &lds_bc[wave][t & 1][0];  // LDS broadcast
        const float du = dt * ut;
        float y = 0.0f;
        #pragma unroll
        for (int n = 0; n < D_STATE; ++n) {
            const float dA = __expf(dt * Aa[n]);
            h[n] = dA * h[n] + du * bc[n];
            y = fmaf(h[n], bc[32 + n], y);
        }

        const float yv  = y + ut * Dd;
        const float sig = 1.0f / (1.0f + __expf(-zt));
        ygated[cidx] = (bf16_t)(yv * (zt * sig));
    }
}

// ---------------------------------------------------------------------------
// Host-side orchestration (graph-capture safe: only kernel launches on stream)
// ---------------------------------------------------------------------------
extern "C" void kernel_launch(void* const* d_in, const int* in_sizes, int n_in,
                              void* d_out, int out_size, void* d_ws, size_t ws_size,
                              hipStream_t stream) {
    (void)in_sizes; (void)n_in; (void)out_size; (void)ws_size;

    const float* x          = (const float*)d_in[0];
    const float* norm_w     = (const float*)d_in[1];
    const float* in_proj_w  = (const float*)d_in[2];
    const float* conv_w     = (const float*)d_in[3];
    const float* conv_b     = (const float*)d_in[4];
    const float* x_proj_w   = (const float*)d_in[5];
    const float* dt_proj_w  = (const float*)d_in[6];
    const float* dt_proj_b  = (const float*)d_in[7];
    const float* A_log      = (const float*)d_in[8];
    const float* Dv         = (const float*)d_in[9];
    const float* dt_bias    = (const float*)d_in[10];
    const float* out_proj_w = (const float*)d_in[11];
    float* out = (float*)d_out;

    // Workspace carve-out (~150 MB total), 256B aligned slices.
    char* p = (char*)d_ws;
    auto carve = [&](size_t bytes) -> char* {
        char* r = p;
        p += (bytes + 255) & ~(size_t)255;
        return r;
    };
    bf16_t* xn    = (bf16_t*)carve((size_t)NTOK * D_MODEL * 2);
    bf16_t* wIn   = (bf16_t*)carve((size_t)2 * D_INNER * D_MODEL * 2);
    bf16_t* wXp   = (bf16_t*)carve((size_t)XPROJ_N * D_INNER * 2);
    bf16_t* wDt   = (bf16_t*)carve((size_t)D_INNER * DT_RANK * 2);
    bf16_t* wOut  = (bf16_t*)carve((size_t)D_MODEL * D_INNER * 2);
    float*  xin   = (float*) carve((size_t)NTOK * D_INNER * 4);
    float*  zbuf  = (float*) carve((size_t)NTOK * D_INNER * 4);
    bf16_t* xcbf  = (bf16_t*)carve((size_t)NTOK * D_INNER * 2);
    float*  xbc   = (float*) carve((size_t)NTOK * XPROJ_N * 4);
    bf16_t* dtr   = (bf16_t*)carve((size_t)NTOK * DT_RANK * 2);
    float*  delta = (float*) carve((size_t)NTOK * D_INNER * 4);
    bf16_t* yg    = (bf16_t*)carve((size_t)NTOK * D_INNER * 2);

    // 1) Weight conversions to bf16.
    {
        int n;
        n = 2 * D_INNER * D_MODEL;
        convert_f32_bf16<<<(n + 255) / 256, 256, 0, stream>>>(in_proj_w, wIn, n);
        n = XPROJ_N * D_INNER;
        convert_f32_bf16<<<(n + 255) / 256, 256, 0, stream>>>(x_proj_w, wXp, n);
        n = D_INNER * DT_RANK;
        convert_f32_bf16<<<(n + 255) / 256, 256, 0, stream>>>(dt_proj_w, wDt, n);
        n = D_MODEL * D_INNER;
        convert_f32_bf16<<<(n + 255) / 256, 256, 0, stream>>>(out_proj_w, wOut, n);
    }

    // 2) RMSNorm -> bf16.
    rmsnorm_kernel<<<NTOK, 256, 0, stream>>>(x, norm_w, xn);

    // 3) in_proj: [4096,1024] x [1024,4096] -> x_in | z  (WMMA, mode 0).
    gemm_wmma_bf16<<<dim3(NTOK / 256, (2 * D_INNER) / 64), 256, 0, stream>>>(
        xn, wIn, D_MODEL, 2 * D_INNER, 0, nullptr, nullptr, xin, zbuf);

    // 4) Depthwise causal conv (in-place on x_in, emits bf16 copy).
    conv1d_kernel<<<(BSZ * D_INNER) / 256, 256, 0, stream>>>(xin, conv_w, conv_b, xcbf);

    // 5) x_proj: [4096,2048] x [2048,128] -> xbc  (WMMA, mode 1).
    gemm_wmma_bf16<<<dim3(NTOK / 256, XPROJ_N / 64), 256, 0, stream>>>(
        xcbf, wXp, D_INNER, XPROJ_N, 1, nullptr, nullptr, xbc, nullptr);

    // 6) Compact dtr columns to bf16.
    {
        int n = NTOK * DT_RANK;
        extract_dtr<<<(n + 255) / 256, 256, 0, stream>>>(xbc, dtr);
    }

    // 7) dt_proj + softplus(delta + dt_proj_b + dt_bias)  (WMMA, mode 2).
    gemm_wmma_bf16<<<dim3(NTOK / 256, D_INNER / 64), 256, 0, stream>>>(
        dtr, wDt, DT_RANK, D_INNER, 2, dt_proj_b, dt_bias, delta, nullptr);

    // 8) Selective scan + skip + SiLU gating -> bf16 (async LDS staging).
    scan_kernel<<<(BSZ * D_INNER) / 256, 256, 0, stream>>>(
        delta, xin, zbuf, xbc, A_log, Dv, yg);

    // 9) out_proj: [4096,2048] x [2048,1024] -> d_out  (WMMA, mode 3).
    gemm_wmma_bf16<<<dim3(NTOK / 256, D_MODEL / 64), 256, 0, stream>>>(
        yg, wOut, D_INNER, D_MODEL, 3, nullptr, nullptr, out, nullptr);
}
